// Attention_12051678232671
// MI455X (gfx1250) — matrix operand order repaired and checked by reading, not compiled, not run on hardware
//
#include <hip/hip_runtime.h>
#include <cstdint>
#include <cstddef>

// Problem constants (from reference): B=2, H=16, S=2048, D=64
#define BDIM 2
#define HDIM 16
#define SDIM 2048
#define DDIM 64
#define QT   32            // q rows per workgroup
#define NTHREADS 256       // 8 waves (wave32)
#define LDP  2052          // padded LDS row stride in floats (2052 % 64 = 4 -> no bank phase lock)

typedef __attribute__((ext_vector_type(16))) __bf16        v16bf;
typedef __attribute__((ext_vector_type(8)))  float         v8f;
typedef __attribute__((ext_vector_type(8)))  unsigned int  v8u;

// Pack hi-16 of two fp32 (i.e. bf16 truncation) into one dword: ONE v_perm_b32.
// result[15:0] = f0[31:16], result[31:16] = f1[31:16]
__device__ __forceinline__ unsigned pk2(float f0, float f1) {
    return __builtin_amdgcn_perm(__builtin_bit_cast(unsigned, f1),
                                 __builtin_bit_cast(unsigned, f0),
                                 0x07060302u);
}

// A-matrix 16x32 bf16 fragment (ISA 7.12.2), with fold-in scale (exact for 2^-n).
// lane m = lane&15, half = lane>>4. Two contiguous 8-float runs per lane -> b128 loads.
__device__ __forceinline__ v16bf load_A_scaled(const float* __restrict__ src, int ld,
                                               int lane, float scale) {
    const int half = lane >> 4;
    const int m    = lane & 15;
    const float4* r0 = (const float4*)(src + (size_t)m * ld + half * 8);        // K = half*8 .. +7
    const float4* r1 = (const float4*)(src + (size_t)m * ld + 16 + half * 8);   // K = 16+half*8 .. +7
    float4 c0 = r0[0], c1 = r0[1], c2 = r1[0], c3 = r1[1];
    v8u u;
    u[0] = pk2(c0.x * scale, c0.y * scale);
    u[1] = pk2(c0.z * scale, c0.w * scale);
    u[2] = pk2(c1.x * scale, c1.y * scale);
    u[3] = pk2(c1.z * scale, c1.w * scale);
    u[4] = pk2(c2.x * scale, c2.y * scale);
    u[5] = pk2(c2.z * scale, c2.w * scale);
    u[6] = pk2(c3.x * scale, c3.y * scale);
    u[7] = pk2(c3.z * scale, c3.w * scale);
    return __builtin_bit_cast(v16bf, u);
}

// Transposed-B fragment for Q*K^T: B[k][n] = src[n*ld + k]; two 8-float
// runs per lane (K = half*8.., 16+half*8..) -> 4x global_load_b128 + 8 perms.
__device__ __forceinline__ v16bf load_Bt(const float* __restrict__ src, int ld, int lane) {
    const int half = lane >> 4;
    const int n    = lane & 15;
    const float4* r0 = (const float4*)(src + (size_t)n * ld + half * 8);
    const float4* r1 = (const float4*)(src + (size_t)n * ld + 16 + half * 8);
    float4 c0 = r0[0], c1 = r0[1], c2 = r1[0], c3 = r1[1];
    v8u u;
    u[0] = pk2(c0.x, c0.y);  u[1] = pk2(c0.z, c0.w);
    u[2] = pk2(c1.x, c1.y);  u[3] = pk2(c1.z, c1.w);
    u[4] = pk2(c2.x, c2.y);  u[5] = pk2(c2.z, c2.w);
    u[6] = pk2(c3.x, c3.y);  u[7] = pk2(c3.z, c3.w);
    return __builtin_bit_cast(v16bf, u);
}

// B-matrix 32x16 fragment, row-major KxN: B[k][n] = src[k*ld + n]
__device__ __forceinline__ v16bf load_B(const float* __restrict__ src, int ld, int lane) {
    const int half = lane >> 4;
    const int n    = lane & 15;
    v8u u;
#pragma unroll
    for (int v = 0; v < 8; ++v) {
        int k = (v < 4) ? (half * 8 + 2 * v) : (16 + half * 8 + 2 * (v - 4));
        u[v] = pk2(src[(size_t)k * ld + n], src[(size_t)(k + 1) * ld + n]);
    }
    return __builtin_bit_cast(v16bf, u);
}

__global__ void __launch_bounds__(NTHREADS, 1)
attn_fwd_kernel(const float* __restrict__ Q, const float* __restrict__ Kmat,
                const float* __restrict__ V, const unsigned char* __restrict__ mask,
                float* __restrict__ out, float* __restrict__ pattn) {
    extern __shared__ __align__(16) float smem[];        // 32 x LDP fp32 scores (padded)
    float* red  = smem + QT * LDP;                       // 256 floats (row-max partials)
    float* red2 = red + NTHREADS;                        // 256 floats (row-sum partials)

    const int tid  = threadIdx.x;
    const int lane = tid & 31;
    const int wave = tid >> 5;
    const int bh   = blockIdx.y;
    const int b    = bh / HDIM;
    const int q0   = blockIdx.x * QT;

    const float* Qb = Q    + (size_t)bh * SDIM * DDIM;
    const float* Kb = Kmat + (size_t)bh * SDIM * DDIM;
    const float* Vb = V    + (size_t)bh * SDIM * DDIM;
    const unsigned char* Mb = mask + (size_t)b * SDIM * SDIM;

    const int half = lane >> 4;
    const int n    = lane & 15;

    // ---------------- Phase A: raw scores = (Q/8) K^T  -> LDS ------------------
    // Q fragments converted ONCE per wave (scale 2^-3 folded in, exact).
    const v16bf a00 = load_A_scaled(Qb + (size_t)q0 * DDIM,          DDIM, lane, 0.125f);
    const v16bf a01 = load_A_scaled(Qb + (size_t)q0 * DDIM + 32,     DDIM, lane, 0.125f);
    const v16bf a10 = load_A_scaled(Qb + (size_t)(q0 + 16) * DDIM,      DDIM, lane, 0.125f);
    const v16bf a11 = load_A_scaled(Qb + (size_t)(q0 + 16) * DDIM + 32, DDIM, lane, 0.125f);

    for (int ct = wave; ct < SDIM / 16; ct += (NTHREADS / 32)) {
        const float* ksrc = Kb + (size_t)(ct * 16) * DDIM;
        v16bf b0 = load_Bt(ksrc,      DDIM, lane);       // d = 0..31
        v16bf b1 = load_Bt(ksrc + 32, DDIM, lane);       // d = 32..63
        v8f acc0 = {}, acc1 = {};
        acc0 = __builtin_amdgcn_wmma_f32_16x16x32_bf16(false, a00, false, b0, (short)0, acc0, false, false);
        acc0 = __builtin_amdgcn_wmma_f32_16x16x32_bf16(false, a01, false, b1, (short)0, acc0, false, false);
        acc1 = __builtin_amdgcn_wmma_f32_16x16x32_bf16(false, a10, false, b0, (short)0, acc1, false, false);
        acc1 = __builtin_amdgcn_wmma_f32_16x16x32_bf16(false, a11, false, b1, (short)0, acc1, false, false);
        const int kcol = ct * 16 + n;
#pragma unroll
        for (int r = 0; r < 8; ++r) {
            smem[(size_t)(r + 8 * half) * LDP + kcol]      = acc0[r];
            smem[(size_t)(16 + r + 8 * half) * LDP + kcol] = acc1[r];
        }
    }
    __syncthreads();

    // ---------------- Phase B: clip + mask + row softmax (vectorized) ----------
    {
        const int row = tid >> 3;                        // 0..31
        const int j   = tid & 7;                         // 8 threads per row
        float4*       rp4 = (float4*)(smem + (size_t)row * LDP);
        const uchar4* mr4 = (const uchar4*)(Mb + (size_t)(q0 + row) * SDIM);

        float mx = -3.0e38f;
        for (int i = 0; i < SDIM / 32; ++i) {            // 64 iters, float4/uchar4
            const int idx = j + 8 * i;
            float4 s = rp4[idx];
            uchar4 m = mr4[idx];
            s.x = m.x ? fminf(fmaxf(s.x, 1e-9f), 1e9f) : -1e9f;
            s.y = m.y ? fminf(fmaxf(s.y, 1e-9f), 1e9f) : -1e9f;
            s.z = m.z ? fminf(fmaxf(s.z, 1e-9f), 1e9f) : -1e9f;
            s.w = m.w ? fminf(fmaxf(s.w, 1e-9f), 1e9f) : -1e9f;
            rp4[idx] = s;
            mx = fmaxf(mx, fmaxf(fmaxf(s.x, s.y), fmaxf(s.z, s.w)));
        }
        red[row * 8 + j] = mx;
        __syncthreads();

        float rowmax = red[row * 8];
#pragma unroll
        for (int jj = 1; jj < 8; ++jj) rowmax = fmaxf(rowmax, red[row * 8 + jj]);

        float sum = 0.0f;
        for (int i = 0; i < SDIM / 32; ++i) {
            const int idx = j + 8 * i;
            float4 s = rp4[idx];
            s.x = __expf(s.x - rowmax);
            s.y = __expf(s.y - rowmax);
            s.z = __expf(s.z - rowmax);
            s.w = __expf(s.w - rowmax);
            rp4[idx] = s;
            sum += (s.x + s.y) + (s.z + s.w);
        }
        red2[row * 8 + j] = sum;
        __syncthreads();

        float tot = 0.0f;
#pragma unroll
        for (int jj = 0; jj < 8; ++jj) tot += red2[row * 8 + jj];
        const float rn = 1.0f / tot;
        for (int i = 0; i < SDIM / 32; ++i) {
            const int idx = j + 8 * i;
            float4 s = rp4[idx];
            s.x *= rn; s.y *= rn; s.z *= rn; s.w *= rn;
            rp4[idx] = s;
        }
    }
    __syncthreads();

    // ---------------- Stream p_attn block to HBM (coalesced b128) --------------
    {
        float4*       dst = (float4*)(pattn + ((size_t)bh * SDIM + q0) * SDIM);
        const float4* src = (const float4*)smem;
        for (int i = tid; i < QT * SDIM / 4; i += NTHREADS) {
            const int row  = i >> 9;                     // 512 float4 per logical row
            const int col4 = i & 511;
            dst[i] = src[row * (LDP / 4) + col4];        // LDP % 4 == 0
        }
    }

    // ---------------- Phase C: out = P @ V  (P from LDS, bf16 WMMA) ------------
    {
        const int rt = wave >> 2;                        // 0..1
        const int ct = wave & 3;                         // 0..3
        const int m  = lane & 15;
        v8f acc = {};
#pragma unroll 2
        for (int kk = 0; kk < SDIM; kk += 32) {
            // A fragment from LDS P block: two 8-float runs -> ds_load_b128 x4
            const float* arow = smem + (size_t)(rt * 16 + m) * LDP + kk;
            const float4* p0 = (const float4*)(arow + half * 8);
            const float4* p1 = (const float4*)(arow + 16 + half * 8);
            float4 c0 = p0[0], c1 = p0[1], c2 = p1[0], c3 = p1[1];
            v8u au;
            au[0] = pk2(c0.x, c0.y);  au[1] = pk2(c0.z, c0.w);
            au[2] = pk2(c1.x, c1.y);  au[3] = pk2(c1.z, c1.w);
            au[4] = pk2(c2.x, c2.y);  au[5] = pk2(c2.z, c2.w);
            au[6] = pk2(c3.x, c3.y);  au[7] = pk2(c3.z, c3.w);
            v16bf a  = __builtin_bit_cast(v16bf, au);
            v16bf bv = load_B(Vb + (size_t)kk * DDIM + ct * 16, DDIM, lane);
            acc = __builtin_amdgcn_wmma_f32_16x16x32_bf16(false, a, false, bv,
                                                          (short)0, acc, false, false);
        }
#pragma unroll
        for (int r = 0; r < 8; ++r) {
            const int q = q0 + rt * 16 + r + 8 * half;
            out[((size_t)bh * SDIM + q) * DDIM + ct * 16 + n] = acc[r];
        }
    }
}

extern "C" void kernel_launch(void* const* d_in, const int* in_sizes, int n_in,
                              void* d_out, int out_size, void* d_ws, size_t ws_size,
                              hipStream_t stream) {
    (void)in_sizes; (void)n_in; (void)out_size; (void)d_ws; (void)ws_size;

    const float*         Q    = (const float*)d_in[0];
    const float*         K    = (const float*)d_in[1];
    const float*         V    = (const float*)d_in[2];
    const unsigned char* mask = (const unsigned char*)d_in[3];   // jnp.bool_ = 1 byte

    float* out   = (float*)d_out;                                 // [B,H,S,D]
    float* pattn = out + (size_t)BDIM * HDIM * SDIM * DDIM;       // [B,H,S,S]

    const size_t shmem = ((size_t)QT * LDP + 2 * NTHREADS) * sizeof(float);  // ~258.5 KB
    (void)hipFuncSetAttribute(reinterpret_cast<const void*>(attn_fwd_kernel),
                              hipFuncAttributeMaxDynamicSharedMemorySize, (int)shmem);

    dim3 grid(SDIM / QT, BDIM * HDIM);   // 64 x 32 = 2048 workgroups
    dim3 block(NTHREADS);
    attn_fwd_kernel<<<grid, block, shmem, stream>>>(Q, K, V, mask, out, pattn);
}
